// HSI_RWKV_43731357008661
// MI455X (gfx1250) — compile-verified, matrix operand order checked by reading, not run on hardware
//
#include <hip/hip_runtime.h>

// ---------------------------------------------------------------------------
// HSI-RWKV fused block for MI455X (gfx1250, wave32, WMMA bf16 16x16x32)
//
// One workgroup = one (batch, 128-token tile, variant) slab.
// All five GEMMs + three LayerNorms fused; activations staged in LDS
// (~262 KB of the 320 KB WGP budget), weights bf16 in global (L2-resident).
// A-fragments are register-hoisted so LDS traffic is ~1 fragment-read per
// K-step per wave instead of per (n,k).
// ---------------------------------------------------------------------------

typedef __attribute__((ext_vector_type(16))) __bf16 v16bf;
typedef __attribute__((ext_vector_type(8)))  float  v8f;

constexpr int kC    = 128;     // channels
constexpr int kHid  = 512;     // hidden (4*C)
constexpr int kHW   = 16384;   // 128*128 tokens per image
constexpr int kTile = 128;     // tokens per workgroup
constexpr int kThr  = 256;     // 8 waves of 32
constexpr int kSXS  = 129;     // f32 row stride (bank-spread)
constexpr int kSAS  = 136;     // bf16 row stride (mult of 8 -> 16B aligned rows)
constexpr int kSKS  = 520;     // bf16 row stride for k (512 + 8)

__device__ __forceinline__ unsigned short f2bf(float f) {
  unsigned u = __builtin_bit_cast(unsigned, f);
  u = (u + 0x7FFFu + ((u >> 16) & 1u)) >> 16;   // round-to-nearest-even
  return (unsigned short)u;
}

// Load one 16x32 bf16 fragment (A or B operand) from a row-major [rows][stride]
// array. CDNA5 layout: lane = half*16 + m; VGPRs 0..3 hold K = half*8..+7,
// VGPRs 4..7 hold K = 16+half*8..+7  -> exactly two 128-bit loads per lane.
__device__ __forceinline__ v16bf load_frag(const unsigned short* p0, int stride,
                                           int row0, int k0, int lane) {
  const int m = lane & 15, half = lane >> 4;
  const unsigned short* p = p0 + (size_t)(row0 + m) * stride + k0 + half * 8;
  union { uint4 q[2]; v16bf v; } u;
  u.q[0] = *(const uint4*)(p);
  u.q[1] = *(const uint4*)(p + 16);
  return u.v;
}

__device__ __forceinline__ v8f wmma_bf16(v16bf a, v16bf b, v8f c) {
  return __builtin_amdgcn_wmma_f32_16x16x32_bf16(false, a, false, b,
                                                 (short)0, c, false, false);
}

__device__ __forceinline__ float sigmoidf(float x) {
  return 1.0f / (1.0f + __expf(-x));
}

// ---------------------------------------------------------------------------
__global__ __launch_bounds__(kThr) void hsi_rwkv_fused(
    const float* __restrict__ x,
    const float* __restrict__ g0, const float* __restrict__ b0,
    const float* __restrict__ g1, const float* __restrict__ b1,
    const float* __restrict__ g2, const float* __restrict__ b2,
    const unsigned short* __restrict__ Wout,
    const unsigned short* __restrict__ Wwht,
    const unsigned short* __restrict__ Wkey,
    const unsigned short* __restrict__ Wrec,
    const unsigned short* __restrict__ Wval,
    float* __restrict__ out)
{
  extern __shared__ char smem[];
  float*          sX = (float*)smem;                        // [128][129] residual f32
  unsigned short* sA = (unsigned short*)(sX + kTile * kSXS);// [128][136] bf16 act
  unsigned short* sW = sA + kTile * kSAS;                   // [128][136] bf16 wht
  unsigned short* sK = sW + kTile * kSAS;                   // [128][520] bf16 k

  const int wg      = blockIdx.x;          // 1024 WGs: batch(2b) | tile(7b) | variant(1b)
  const int variant = wg & 1;
  const int tile    = (wg >> 1) & 127;
  const int batch   = wg >> 8;
  const int t0      = tile * kTile;
  const float* xb = x   + (size_t)batch * kC * kHW;
  float*       ob = out + (size_t)batch * kC * kHW;

  const int tid  = threadIdx.x;
  const int wave = tid >> 5;
  const int lane = tid & 31;
  const int lm   = lane & 15, lh = lane >> 4;
  const int m0   = wave * 16;

  // Warm L2 with the two big weight matrices (emits global_prefetch_b8).
  __builtin_prefetch(Wkey + (size_t)tid * 256, 0, 1);
  __builtin_prefetch(Wval + (size_t)tid * 256, 0, 1);

  // ---- load 128 tokens x 128 channels (coalesced over token dim) ----------
  for (int i = tid; i < kTile * kC; i += kThr) {
    const int tok = i & (kTile - 1);
    const int ch  = i >> 7;
    const int tt  = t0 + tok;
    const int off = variant ? (((tt & 127) << 7) | (tt >> 7)) : tt;
    sX[tok * kSXS + ch] = xb[(size_t)ch * kHW + off];
  }
  __syncthreads();

  // ---- LN0 (in place on residual stream) ----------------------------------
  if (tid < kTile) {
    float* row = sX + tid * kSXS;
    float mu = 0.f;
    for (int c = 0; c < kC; ++c) mu += row[c];
    mu *= (1.0f / kC);
    float var = 0.f;
    for (int c = 0; c < kC; ++c) { float d = row[c] - mu; var += d * d; }
    float rs = rsqrtf(var * (1.0f / kC) + 1e-5f);
    for (int c = 0; c < kC; ++c) row[c] = (row[c] - mu) * rs * g0[c] + b0[c];
  }
  __syncthreads();

  // ---- LN1 + SiLU gate -> sA (bf16) ---------------------------------------
  if (tid < kTile) {
    const float* row = sX + tid * kSXS;
    float mu = 0.f;
    for (int c = 0; c < kC; ++c) mu += row[c];
    mu *= (1.0f / kC);
    float var = 0.f;
    for (int c = 0; c < kC; ++c) { float d = row[c] - mu; var += d * d; }
    float rs = rsqrtf(var * (1.0f / kC) + 1e-5f);
    unsigned short* dst = sA + tid * kSAS;
    for (int c = 0; c < kC; ++c) {
      float y = (row[c] - mu) * rs * g1[c] + b1[c];
      dst[c] = f2bf(sigmoidf(y) * y);
    }
  }
  __syncthreads();

  // ---- GEMM1: att = gate @ Wout^T, residual-add into sX -------------------
  {
    v16bf a[4];
    #pragma unroll
    for (int k = 0; k < 4; ++k) a[k] = load_frag(sA, kSAS, m0, k * 32, lane);
    for (int n = 0; n < 8; ++n) {
      v8f acc = {0.f,0.f,0.f,0.f,0.f,0.f,0.f,0.f};
      #pragma unroll
      for (int k = 0; k < 4; ++k)
        acc = wmma_bf16(a[k], load_frag(Wout, kC, n * 16, k * 32, lane), acc);
      #pragma unroll
      for (int v = 0; v < 8; ++v)
        sX[(m0 + v + lh * 8) * kSXS + n * 16 + lm] += acc[v];
    }
  }
  __syncthreads();

  // ---- LN2 -> sA (bf16) ----------------------------------------------------
  if (tid < kTile) {
    const float* row = sX + tid * kSXS;
    float mu = 0.f;
    for (int c = 0; c < kC; ++c) mu += row[c];
    mu *= (1.0f / kC);
    float var = 0.f;
    for (int c = 0; c < kC; ++c) { float d = row[c] - mu; var += d * d; }
    float rs = rsqrtf(var * (1.0f / kC) + 1e-5f);
    unsigned short* dst = sA + tid * kSAS;
    for (int c = 0; c < kC; ++c)
      dst[c] = f2bf((row[c] - mu) * rs * g2[c] + b2[c]);
  }
  __syncthreads();

  // ---- GEMM2: wht = y2 @ Wwht^T -> sW (bf16) ------------------------------
  {
    v16bf a[4];
    #pragma unroll
    for (int k = 0; k < 4; ++k) a[k] = load_frag(sA, kSAS, m0, k * 32, lane);
    for (int n = 0; n < 8; ++n) {
      v8f acc = {0.f,0.f,0.f,0.f,0.f,0.f,0.f,0.f};
      #pragma unroll
      for (int k = 0; k < 4; ++k)
        acc = wmma_bf16(a[k], load_frag(Wwht, kC, n * 16, k * 32, lane), acc);
      #pragma unroll
      for (int v = 0; v < 8; ++v)
        sW[(m0 + v + lh * 8) * kSAS + n * 16 + lm] = f2bf(acc[v]);
    }
  }
  __syncthreads();

  // ---- GEMM3: k = relu(wht @ Wkey^T)^2 -> sK (bf16, 128x512) --------------
  {
    v16bf a[4];
    #pragma unroll
    for (int k = 0; k < 4; ++k) a[k] = load_frag(sW, kSAS, m0, k * 32, lane);
    for (int n = 0; n < 32; ++n) {
      v8f acc = {0.f,0.f,0.f,0.f,0.f,0.f,0.f,0.f};
      #pragma unroll
      for (int k = 0; k < 4; ++k)
        acc = wmma_bf16(a[k], load_frag(Wkey, kC, n * 16, k * 32, lane), acc);
      #pragma unroll
      for (int v = 0; v < 8; ++v) {
        float r = acc[v] > 0.f ? acc[v] : 0.f;
        sK[(m0 + v + lh * 8) * kSKS + n * 16 + lm] = f2bf(r * r);
      }
    }
  }
  __syncthreads();

  // ---- GEMM4: kv = k @ Wval^T (K=512), k-outer so each sK fragment is
  //      loaded from LDS exactly once; 8 accumulator tiles stay live ---------
  {
    v8f kv[8];
    #pragma unroll
    for (int n = 0; n < 8; ++n) kv[n] = (v8f){0.f,0.f,0.f,0.f,0.f,0.f,0.f,0.f};
    for (int k = 0; k < 16; ++k) {
      v16bf a = load_frag(sK, kSKS, m0, k * 32, lane);
      #pragma unroll
      for (int n = 0; n < 8; ++n)
        kv[n] = wmma_bf16(a, load_frag(Wval, kHid, n * 16, k * 32, lane), kv[n]);
    }

    // ---- GEMM5: r = sigmoid(wht @ Wrec^T), fused epilogue:
    //      out += sX + r*kv   (atomic across the two variants) --------------
    v16bf aw[4];
    #pragma unroll
    for (int k = 0; k < 4; ++k) aw[k] = load_frag(sW, kSAS, m0, k * 32, lane);
    for (int n = 0; n < 8; ++n) {
      v8f rr = {0.f,0.f,0.f,0.f,0.f,0.f,0.f,0.f};
      #pragma unroll
      for (int k = 0; k < 4; ++k)
        rr = wmma_bf16(aw[k], load_frag(Wrec, kC, n * 16, k * 32, lane), rr);
      #pragma unroll
      for (int v = 0; v < 8; ++v) {
        const int row = m0 + v + lh * 8;
        const int col = n * 16 + lm;
        const float val = sX[row * kSXS + col] + sigmoidf(rr[v]) * kv[n][v];
        const int tt  = t0 + row;
        const int off = variant ? (((tt & 127) << 7) | (tt >> 7)) : tt;
        atomicAdd(ob + (size_t)col * kHW + off, val);
      }
    }
  }
}

// ---------------------------------------------------------------------------
__global__ void f32_to_bf16(const float* __restrict__ s,
                            unsigned short* __restrict__ d, int n) {
  int i = blockIdx.x * blockDim.x + threadIdx.x;
  if (i < n) d[i] = f2bf(s[i]);
}

__global__ void zero_f32(float* __restrict__ p, int n) {
  int i = blockIdx.x * blockDim.x + threadIdx.x;
  if (i < n) p[i] = 0.f;
}

// ---------------------------------------------------------------------------
extern "C" void kernel_launch(void* const* d_in, const int* in_sizes, int n_in,
                              void* d_out, int out_size, void* d_ws, size_t ws_size,
                              hipStream_t stream) {
  const float* inp  = (const float*)d_in[0];
  const float* g0   = (const float*)d_in[1];
  const float* b0   = (const float*)d_in[2];
  const float* g1   = (const float*)d_in[3];
  const float* b1   = (const float*)d_in[4];
  const float* g2   = (const float*)d_in[5];
  const float* b2   = (const float*)d_in[6];
  const float* WoF  = (const float*)d_in[7];
  const float* WwF  = (const float*)d_in[8];
  const float* WkF  = (const float*)d_in[9];
  const float* WrF  = (const float*)d_in[10];
  const float* WvF  = (const float*)d_in[11];

  unsigned short* Wout = (unsigned short*)d_ws;   // bf16 weight staging in ws
  unsigned short* Wwht = Wout + kC * kC;
  unsigned short* Wkey = Wwht + kC * kC;
  unsigned short* Wrec = Wkey + kHid * kC;
  unsigned short* Wval = Wrec + kC * kC;

  f32_to_bf16<<<(kC * kC   + 255) / 256, 256, 0, stream>>>(WoF, Wout, kC * kC);
  f32_to_bf16<<<(kC * kC   + 255) / 256, 256, 0, stream>>>(WwF, Wwht, kC * kC);
  f32_to_bf16<<<(kHid * kC + 255) / 256, 256, 0, stream>>>(WkF, Wkey, kHid * kC);
  f32_to_bf16<<<(kC * kC   + 255) / 256, 256, 0, stream>>>(WrF, Wrec, kC * kC);
  f32_to_bf16<<<(kC * kHid + 255) / 256, 256, 0, stream>>>(WvF, Wval, kC * kHid);

  zero_f32<<<(out_size + 255) / 256, 256, 0, stream>>>((float*)d_out, out_size);

  const size_t smem = (size_t)kTile * kSXS * 4      // sX f32
                    + (size_t)kTile * kSAS * 2 * 2  // sA + sW bf16
                    + (size_t)kTile * kSKS * 2;     // sK bf16  (~262 KB)
  hipFuncSetAttribute((const void*)hsi_rwkv_fused,
                      hipFuncAttributeMaxDynamicSharedMemorySize, (int)smem);

  // 4 batches * 128 token-tiles * 2 variants = 1024 workgroups
  hsi_rwkv_fused<<<1024, kThr, smem, stream>>>(
      inp, g0, b0, g1, b1, g2, b2, Wout, Wwht, Wkey, Wrec, Wval, (float*)d_out);
}
